// DynamicCapsules_7627861918152
// MI455X (gfx1250) — matrix operand order repaired and checked by reading, not compiled, not run on hardware
//
#include <hip/hip_runtime.h>
#include <hip/hip_bf16.h>

typedef __attribute__((ext_vector_type(16))) _Float16 v16h;
typedef __attribute__((ext_vector_type(8)))  _Float16 v8h;
typedef __attribute__((ext_vector_type(8)))  float    v8f;
typedef __attribute__((ext_vector_type(4)))  int      v4i;

#define CAP_EPS 1e-8f
#define BATCH 64
#define H1    56
#define NPRIM 18432     // 32 caps * 24*24
#define KDIM  20736     // 81 * 256
#define MROWS 36864     // 64 * 576
#define KSTEPS 648      // KDIM / 32

#if defined(__gfx1250__) && __has_builtin(__builtin_amdgcn_global_load_async_to_lds_b128)
#define ASYNC_OK 1
#endif

__device__ __forceinline__ void async_b128(const _Float16* g, _Float16* l)
{
#ifdef ASYNC_OK
    __builtin_amdgcn_global_load_async_to_lds_b128(
        (__attribute__((address_space(1))) v4i*)g,
        (__attribute__((address_space(3))) v4i*)l, 0, 0);
#else
    *(v8h*)l = *(const v8h*)g;
#endif
}

__device__ __forceinline__ void wait_async4()
{
#ifdef ASYNC_OK
# if __has_builtin(__builtin_amdgcn_s_wait_asynccnt)
    __builtin_amdgcn_s_wait_asynccnt(4);
# else
    asm volatile("s_wait_asynccnt 0x4" ::: "memory");
# endif
#endif
}

__device__ __forceinline__ void wait_async0()
{
#ifdef ASYNC_OK
# if __has_builtin(__builtin_amdgcn_s_wait_asynccnt)
    __builtin_amdgcn_s_wait_asynccnt(0);
# else
    asm volatile("s_wait_asynccnt 0x0" ::: "memory");
# endif
#endif
}

// ---------------- conv1: 1->256, 9x9, s1, ReLU; output NHWC f16 ----------------
__global__ __launch_bounds__(256)
void k_conv1(const float* __restrict__ x, const float* __restrict__ w,
             const float* __restrict__ bias, _Float16* __restrict__ act)
{
    __shared__ float xl[9 * 64];
    const int tid = threadIdx.x;
    const int b   = blockIdx.x / H1;
    const int oy  = blockIdx.x % H1;
    for (int t = tid; t < 9 * 64; t += 256)
        xl[t] = x[b * 4096 + (oy + (t >> 6)) * 64 + (t & 63)];
    __syncthreads();

    const int ch = tid;
    const float bv = bias[ch];
    const float* wp = w + ch * 81;
    for (int ox0 = 0; ox0 < H1; ox0 += 14) {
        float acc[14];
        #pragma unroll
        for (int j = 0; j < 14; ++j) acc[j] = bv;
        for (int kh = 0; kh < 9; ++kh) {
            #pragma unroll
            for (int kw = 0; kw < 9; ++kw) {
                const float wv = wp[kh * 9 + kw];
                const float* row = &xl[kh * 64 + ox0 + kw];
                #pragma unroll
                for (int j = 0; j < 14; ++j) acc[j] += wv * row[j];
            }
        }
        #pragma unroll
        for (int j = 0; j < 14; ++j) {
            float r = acc[j] > 0.f ? acc[j] : 0.f;
            act[(((size_t)(b * H1 + oy) * H1) + ox0 + j) * 256 + ch] = (_Float16)r;
        }
    }
}

// ---------------- weight prep: (O,I,9,9) f32 -> Bmat[o][ (kh*9+kw)*256 + c ] f16 ----------------
__global__ void k_wprep(const float* __restrict__ pw, _Float16* __restrict__ Bm)
{
    const int idx = blockIdx.x * 256 + threadIdx.x;   // 256*20736 exact
    const int o   = idx / KDIM;
    const int r   = idx % KDIM;
    const int tap = r >> 8;
    const int c   = r & 255;
    Bm[idx] = (_Float16)pw[(o * 256 + c) * 81 + tap];
}

// ---------------- implicit-GEMM pconv via WMMA f16 -> f32 ----------------
// M=36864 (b,oy,ox), N=256, K=20736; WG tile 128x128, wave tile 32x64.
// Double-buffered LDS fed by GLOBAL_LOAD_ASYNC_TO_LDS_B128 (ASYNCcnt).
__global__ __launch_bounds__(256)
void k_gemm(const _Float16* __restrict__ A, const _Float16* __restrict__ Bm,
            float* __restrict__ C)
{
    __shared__ __align__(16) _Float16 Alds[2][128 * 32];   // 2 x 8 KB
    __shared__ __align__(16) _Float16 Blds[2][128 * 32];   // 2 x 8 KB
    const int tid  = threadIdx.x;
    const int lane = tid & 31;
    const int wave = tid >> 5;
    const int wm   = (wave & 3) * 32;    // 4 waves along M
    const int wn   = (wave >> 2) * 64;   // 2 waves along N
    const int Mbase = blockIdx.x * 128;
    const int Nbase = blockIdx.y * 128;

    // staging: both tiles are 128 rows x 32 halfs; thread -> (row, 16-half segment)
    const int row = tid >> 1;
    const int seg = (tid & 1) * 16;
    const int m  = Mbase + row;
    const int b  = m / 576;
    const int yx = m % 576;
    const int oy = yx / 24;
    const int ox = yx % 24;
    const _Float16* Brow = Bm + (size_t)(Nbase + row) * KDIM + seg;

    const int g  = lane >> 4;   // lane group (ISA 16-bit fragment layout)
    const int sl = lane & 15;

    auto issue = [&](int st, int buf) {
        const int kc  = st * 32;
        const int tap = kc >> 8;
        const int c0  = kc & 255;
        const int kh  = tap / 9, kw = tap % 9;
        const int iy  = 2 * oy + kh, ix = 2 * ox + kw;
        const _Float16* Ap = A + (((size_t)((b * 56 + iy) * 56 + ix)) << 8) + c0 + seg;
        async_b128(Ap,     &Alds[buf][row * 32 + seg]);
        async_b128(Ap + 8, &Alds[buf][row * 32 + seg + 8]);
        const _Float16* Bp = Brow + kc;
        async_b128(Bp,     &Blds[buf][row * 32 + seg]);
        async_b128(Bp + 8, &Blds[buf][row * 32 + seg + 8]);
#ifndef ASYNC_OK
        __builtin_prefetch(Bp + 32, 0, 0);
#endif
    };

    v8f zero = {};
    v8f acc[2][4];
    #pragma unroll
    for (int mi = 0; mi < 2; ++mi)
        #pragma unroll
        for (int ni = 0; ni < 4; ++ni) acc[mi][ni] = zero;

    issue(0, 0);
    for (int st = 0; st < KSTEPS; ++st) {
        const int buf = st & 1;
        if (st + 1 < KSTEPS) { issue(st + 1, buf ^ 1); wait_async4(); }
        else                 { wait_async0(); }
        __syncthreads();    // stage-buf data visible to all waves

        // A fragment (16x32 f16): lane halfs [g*8 .. +7] and [16+g*8 .. +7]
        v16h af[2], bf[4];
        #pragma unroll
        for (int mi = 0; mi < 2; ++mi) {
            const _Float16* p = &Alds[buf][(wm + mi * 16 + sl) * 32 + g * 8];
            ((v8h*)&af[mi])[0] = *(const v8h*)p;
            ((v8h*)&af[mi])[1] = *(const v8h*)(p + 16);
        }
        // B fragment (32x16 f16): lane halfs K = [g*8 .. +7] and [16+g*8 .. +7] of column n
        #pragma unroll
        for (int ni = 0; ni < 4; ++ni) {
            const _Float16* p = &Blds[buf][(wn + ni * 16 + sl) * 32 + g * 8];
            ((v8h*)&bf[ni])[0] = *(const v8h*)p;
            ((v8h*)&bf[ni])[1] = *(const v8h*)(p + 16);
        }
        #pragma unroll
        for (int mi = 0; mi < 2; ++mi)
            #pragma unroll
            for (int ni = 0; ni < 4; ++ni)
                acc[mi][ni] = __builtin_amdgcn_wmma_f32_16x16x32_f16(
                    false, af[mi], false, bf[ni], (short)0, acc[mi][ni], false, false);
        __syncthreads();    // all reads of buf done before it is refilled
    }

    // C layout: VGPR r -> M = r + 8*g, N = sl
    #pragma unroll
    for (int mi = 0; mi < 2; ++mi)
        #pragma unroll
        for (int ni = 0; ni < 4; ++ni) {
            const int ncol = Nbase + wn + ni * 16 + sl;
            #pragma unroll
            for (int r = 0; r < 8; ++r) {
                const int mrow = Mbase + wm + mi * 16 + g * 8 + r;
                C[(size_t)mrow * 256 + ncol] = acc[mi][ni][r];
            }
        }
}

// ---------------- squash primary caps: gemm(b,pix,ch)+bias -> u[b][caps*576+pix][8] ----------------
__global__ void k_squash(const float* __restrict__ gm, const float* __restrict__ bias,
                         float* __restrict__ u)
{
    const int idx  = blockIdx.x * 256 + threadIdx.x;  // 64*32*576 exact
    const int b    = idx / (32 * 576);
    const int r    = idx % (32 * 576);
    const int caps = r / 576;
    const int pix  = r % 576;
    const float* gp = gm + ((size_t)(b * 576 + pix)) * 256 + caps * 8;
    float v[8]; float sq = 0.f;
    #pragma unroll
    for (int d = 0; d < 8; ++d) { v[d] = gp[d] + bias[caps * 8 + d]; sq += v[d] * v[d]; }
    const float f = sq / (1.f + sq) / (sqrtf(sq) + CAP_EPS);
    float* up = u + ((size_t)b * NPRIM + caps * 576 + pix) * 8;
    #pragma unroll
    for (int d = 0; d < 8; ++d) up[d] = v[d] * f;
}

// ---------------- u_hat[b][k][n][D] = sum_d W[k][n][D][d]*u[b][n][d] (W stays in L2) ----------------
__global__ void k_uhat(const float* __restrict__ W, const float* __restrict__ u,
                       float* __restrict__ uh)
{
    const size_t idx = (size_t)blockIdx.x * 256 + threadIdx.x;  // 75,497,472 exact
    const int D  = (int)(idx & 15);
    const size_t t = idx >> 4;
    const int n  = (int)(t % NPRIM);
    const int t2 = (int)(t / NPRIM);
    const int k  = t2 & 3;
    const int b  = t2 >> 2;
    const float* wp = W + (((size_t)k * NPRIM + n) * 16 + D) * 8;
    const float* up = u + ((size_t)b * NPRIM + n) * 8;
    float acc = 0.f;
    #pragma unroll
    for (int d = 0; d < 8; ++d) acc += wp[d] * up[d];
    uh[idx] = acc;
}

__global__ void k_zero(float* __restrict__ p)
{
    p[(size_t)blockIdx.x * 256 + threadIdx.x] = 0.f;
}

// ---------------- routing: s = sum_n softmax_k(b)*u_hat; optional squash -> v[b][k][16] ----------------
__global__ __launch_bounds__(256)
void k_sdot(const float* __restrict__ uh, const float* __restrict__ blog,
            float* __restrict__ v, int do_squash)
{
    __shared__ float sh[256 * 16];
    const int tid = threadIdx.x;
    const int b   = blockIdx.x >> 2;
    const int k   = blockIdx.x & 3;
    float acc[16];
    #pragma unroll
    for (int d = 0; d < 16; ++d) acc[d] = 0.f;
    const float* uhb = uh + ((size_t)(b * 4 + k)) * NPRIM * 16;
    const float* lb  = blog + (size_t)b * 4 * NPRIM;
    for (int n = tid; n < NPRIM; n += 256) {
        const float l0 = lb[0 * NPRIM + n], l1 = lb[1 * NPRIM + n];
        const float l2 = lb[2 * NPRIM + n], l3 = lb[3 * NPRIM + n];
        const float mx = fmaxf(fmaxf(l0, l1), fmaxf(l2, l3));
        const float e0 = __expf(l0 - mx), e1 = __expf(l1 - mx);
        const float e2 = __expf(l2 - mx), e3 = __expf(l3 - mx);
        const float ek = (k == 0) ? e0 : (k == 1) ? e1 : (k == 2) ? e2 : e3;
        const float c  = ek / (e0 + e1 + e2 + e3);
        const float* up = uhb + (size_t)n * 16;
        #pragma unroll
        for (int d = 0; d < 16; ++d) acc[d] += c * up[d];
    }
    #pragma unroll
    for (int d = 0; d < 16; ++d) sh[tid * 16 + d] = acc[d];
    __syncthreads();
    for (int s = 128; s > 0; s >>= 1) {
        if (tid < s) {
            #pragma unroll
            for (int d = 0; d < 16; ++d) sh[tid * 16 + d] += sh[(tid + s) * 16 + d];
        }
        __syncthreads();
    }
    if (tid == 0) {
        float sq = 0.f;
        for (int d = 0; d < 16; ++d) sq += sh[d] * sh[d];
        const float f = do_squash ? (sq / (1.f + sq) / (sqrtf(sq) + CAP_EPS)) : 1.f;
        for (int d = 0; d < 16; ++d) v[(b * 4 + k) * 16 + d] = sh[d] * f;
    }
}

// ---------------- b += <u_hat, v> ----------------
__global__ void k_bup(const float* __restrict__ uh, const float* __restrict__ v,
                      float* __restrict__ blog)
{
    const size_t idx = (size_t)blockIdx.x * 256 + threadIdx.x;  // 4,718,592 exact
    const int t = (int)(idx / NPRIM);   // b*4+k
    const float* up = uh + idx * 16;
    const float* vp = v + t * 16;
    float acc = 0.f;
    #pragma unroll
    for (int d = 0; d < 16; ++d) acc += up[d] * vp[d];
    blog[idx] += acc;
}

// ---------------- final FC: out[b][k][o] = <v, fc_w[o]> + fc_b[o] ----------------
__global__ void k_fc(const float* __restrict__ v, const float* __restrict__ fw,
                     const float* __restrict__ fb, float* __restrict__ out)
{
    const int idx = blockIdx.x * 256 + threadIdx.x;
    if (idx >= BATCH * 4 * 10) return;
    const int o = idx % 10;
    const int t = idx / 10;
    float acc = fb[o];
    #pragma unroll
    for (int D = 0; D < 16; ++D) acc += v[t * 16 + D] * fw[o * 16 + D];
    out[idx] = acc;
}

extern "C" void kernel_launch(void* const* d_in, const int* in_sizes, int n_in,
                              void* d_out, int out_size, void* d_ws, size_t ws_size,
                              hipStream_t stream) {
    const float* x   = (const float*)d_in[0];
    const float* c1w = (const float*)d_in[1];
    const float* c1b = (const float*)d_in[2];
    const float* pw  = (const float*)d_in[3];
    const float* pb  = (const float*)d_in[4];
    const float* Wr  = (const float*)d_in[5];
    const float* fw  = (const float*)d_in[6];
    const float* fb  = (const float*)d_in[7];
    float* out = (float*)d_out;
    (void)in_sizes; (void)n_in; (void)out_size; (void)ws_size;

    char* ws = (char*)d_ws;
    size_t off = 0;
    auto carve = [&](size_t bytes) -> void* {
        void* p = ws + off;
        off += (bytes + 255) & ~(size_t)255;
        return p;
    };
    _Float16* act = (_Float16*)carve((size_t)BATCH * H1 * H1 * 256 * 2);  // 102.8 MB
    _Float16* Bm  = (_Float16*)carve((size_t)256 * KDIM * 2);             // 10.6 MB
    float* gemm   = (float*)carve((size_t)MROWS * 256 * 4);               // 37.7 MB
    float* u      = (float*)carve((size_t)BATCH * NPRIM * 8 * 4);         // 37.7 MB
    float* uh     = (float*)carve((size_t)BATCH * 4 * NPRIM * 16 * 4);    // 302 MB
    float* blog   = (float*)carve((size_t)BATCH * 4 * NPRIM * 4);         // 18.9 MB
    float* vv     = (float*)carve((size_t)BATCH * 4 * 16 * 4);            // 16 KB

    k_conv1 <<<dim3(BATCH * H1),   dim3(256), 0, stream>>>(x, c1w, c1b, act);
    k_wprep <<<dim3(20736),        dim3(256), 0, stream>>>(pw, Bm);
    k_gemm  <<<dim3(288, 2),       dim3(256), 0, stream>>>(act, Bm, gemm);
    k_squash<<<dim3(4608),         dim3(256), 0, stream>>>(gemm, pb, u);
    k_uhat  <<<dim3(294912),       dim3(256), 0, stream>>>(Wr, u, uh);
    k_zero  <<<dim3(18432),        dim3(256), 0, stream>>>(blog);
    for (int it = 0; it < 2; ++it) {
        k_sdot<<<dim3(256),  dim3(256), 0, stream>>>(uh, blog, vv, 1);
        k_bup <<<dim3(18432),dim3(256), 0, stream>>>(uh, vv, blog);
    }
    k_sdot<<<dim3(256), dim3(256), 0, stream>>>(uh, blog, vv, 0);
    k_fc  <<<dim3(10),  dim3(256), 0, stream>>>(vv, fw, fb, out);
}
